// BigBirdBlockSparseAttention_17231408791599
// MI455X (gfx1250) — matrix operand order repaired and checked by reading, not compiled, not run on hardware
//
#include <hip/hip_runtime.h>
#include <hip/hip_bf16.h>
#include <math.h>

// ---------------------------------------------------------------------------
// BigBird block-sparse attention for MI455X (gfx1250, wave32, WMMA bf16, TDM)
// B=4, S=4096, H=16, HS=1024, D=64, BLK=64, nb=64
// ---------------------------------------------------------------------------

typedef __bf16 bf16_t;
typedef __bf16 v16bf __attribute__((ext_vector_type(16)));
typedef __bf16 v8bf  __attribute__((ext_vector_type(8)));
typedef float v8f    __attribute__((ext_vector_type(8)));
typedef unsigned int v4u  __attribute__((ext_vector_type(4)));
typedef int          v8i_t __attribute__((ext_vector_type(8)));
typedef int          v4i_t __attribute__((ext_vector_type(4)));

#define BB   4
#define SS   4096
#define HH   16
#define HSZ  1024
#define DD   64
#define NB   64
#define ATT_SCALE 0.125f   // 1/sqrt(64)

#if __has_builtin(__builtin_amdgcn_tensor_load_to_lds)
#define HAVE_TDM 1
#else
#define HAVE_TDM 0
#endif

union F16U { v16bf v; v8bf h[2]; };

__device__ __forceinline__ v8f wmma_bf16(const v16bf& a, const v16bf& b, const v8f& c) {
  return __builtin_amdgcn_wmma_f32_16x16x32_bf16(false, a, false, b, (short)0, c, false, false);
}

// xor-shuffle within 32 lanes via ds_swizzle (group-of-32 mode, and=0x1f)
template <int M>
__device__ __forceinline__ float swz_xor(float v) {
  constexpr int imm = (M << 10) | 0x1f;
  return __int_as_float(__builtin_amdgcn_ds_swizzle(__float_as_int(v), imm));
}
__device__ __forceinline__ float rmax16(float v) {
  v = fmaxf(v, swz_xor<1>(v)); v = fmaxf(v, swz_xor<2>(v));
  v = fmaxf(v, swz_xor<4>(v)); v = fmaxf(v, swz_xor<8>(v));
  return v;
}
__device__ __forceinline__ float rsum16(float v) {
  v += swz_xor<1>(v); v += swz_xor<2>(v);
  v += swz_xor<4>(v); v += swz_xor<8>(v);
  return v;
}

#if HAVE_TDM
// Tensor Data Mover: DMA a 2D tile of bf16 (tile_w contiguous elements per row,
// tile_h rows, row stride stride_elems) from global memory into LDS.
// D# layout per CDNA5 ISA (groups 0/1; groups 2/3 zero,
// tile_dim2=0 => 2D). Tracked by TENSORcnt.
// This toolchain's builtin takes 6 args (clang-23 / therock-10.0 form):
//   (uint32x4 g0, int32x8 g1, int32x4 g2, int32x4 g3, int32x8, i32 cpol)
__device__ __forceinline__ void tdm_load_2d_bf16(unsigned lds_off, const void* gptr,
                                                 unsigned tile_w, unsigned tile_h,
                                                 unsigned stride_elems) {
  unsigned long long ga = (unsigned long long)(uintptr_t)gptr;
  v4u g0;
  g0[0] = 1u;                                            // count=1, user mode
  g0[1] = lds_off;                                       // lds_addr (bytes)
  g0[2] = (unsigned)ga;                                  // global_addr[31:0]
  g0[3] = (unsigned)((ga >> 32) & 0x01ffffffu) | (2u << 30);  // addr[56:32], type=2
  v8i_t g1;
  g1[0] = (int)(1u << 16);                               // data_size=1 => 2 bytes
  g1[1] = (int)(tile_w << 16);                           // tensor_dim0[15:0] @ bit48
  g1[2] = (int)((tile_w >> 16) | (tile_h << 16));        // tdim0[31:16] | tdim1[15:0]
  g1[3] = (int)((tile_h >> 16) | (tile_w << 16));        // tdim1[31:16] | tile_dim0
  g1[4] = (int)tile_h;                                   // tile_dim1 (tile_dim2=0)
  g1[5] = (int)stride_elems;                             // tensor_dim0_stride[31:0]
  g1[6] = 0;                                             // stride[47:32] | tdim1_stride lo
  g1[7] = 0;
  v4i_t z4 = {0, 0, 0, 0};
  v8i_t z8 = {0, 0, 0, 0, 0, 0, 0, 0};
  __builtin_amdgcn_tensor_load_to_lds(g0, g1, z4, z4, z8, 0);
}
#endif

__device__ __forceinline__ int kb_of(int l, int i) {
  if (l == 0 || l == NB - 1) return i;                        // full rows
  if (l == 1)       return (i < 3) ? i : (NB - 1);            // {0,1,2,63}
  if (l == NB - 2)  return (i == 0) ? 0 : (NB - 4 + i);       // {0,61,62,63}
  return (i == 0) ? 0 : ((i == 4) ? (NB - 1) : (l - 2 + i));  // {0,l-1,l,l+1,63}
}

// ---------------------------------------------------------------------------
// Kernel 1: fp32 hidden_states -> bf16, 8 elements per thread
// ---------------------------------------------------------------------------
__global__ __launch_bounds__(256) void convert_x_kernel(
    const float* __restrict__ X, bf16_t* __restrict__ Xb, int n8) {
  int i = blockIdx.x * blockDim.x + threadIdx.x;
  if (i >= n8) return;
  const float4* p = reinterpret_cast<const float4*>(X) + (size_t)i * 2;
  float4 a = p[0], b = p[1];
  v8bf o;
  o[0] = (bf16_t)a.x; o[1] = (bf16_t)a.y; o[2] = (bf16_t)a.z; o[3] = (bf16_t)a.w;
  o[4] = (bf16_t)b.x; o[5] = (bf16_t)b.y; o[6] = (bf16_t)b.z; o[7] = (bf16_t)b.w;
  *reinterpret_cast<v8bf*>(Xb + (size_t)i * 8) = o;
}

// ---------------------------------------------------------------------------
// Kernel 2: W[n][k] fp32 -> Wbt[t][k][n] bf16 (K-major => WMMA B-frag rows
// contiguous)
// ---------------------------------------------------------------------------
__global__ __launch_bounds__(256) void transpose_w_kernel(
    const float* __restrict__ Wq, const float* __restrict__ Wk,
    const float* __restrict__ Wv, bf16_t* __restrict__ Wbt) {
  int tid = blockIdx.x * blockDim.x + threadIdx.x;      // < 3 * 2^20
  int t = tid >> 20;
  int r = tid & ((1 << 20) - 1);
  int k = r >> 10;
  int n = r & 1023;
  const float* W = (t == 0) ? Wq : (t == 1) ? Wk : Wv;
  Wbt[tid] = (bf16_t)W[(size_t)n * HSZ + k];
}

// ---------------------------------------------------------------------------
// Kernel 3: fused QKV projection GEMM.
//   C[r][n] = sum_k Xb[r][k] * W_t[n][k] + bias_t[n]   (r = b*S + s)
// grid = (M/128, N/64, 3), block = 128 (4 waves), each wave a 32x64 strip
// (8 WMMAs per K-step vs 4 b128 A loads + 64 16-bit B loads).
// Q/V stored (B,H,S,64) bf16; K stored transposed (B,H,64,S) bf16.
// ---------------------------------------------------------------------------
__global__ __launch_bounds__(128) void qkv_gemm_kernel(
    const bf16_t* __restrict__ Xb, const bf16_t* __restrict__ Wbt,
    const float* __restrict__ bq, const float* __restrict__ bk,
    const float* __restrict__ bv,
    bf16_t* __restrict__ Q, bf16_t* __restrict__ Kt, bf16_t* __restrict__ V) {
  const int t    = blockIdx.z;
  const int m0   = blockIdx.x * 128;
  const int n0   = blockIdx.y * 64;
  const int wave = threadIdx.x >> 5;
  const int lane = threadIdx.x & 31;
  const int lhalf = lane >> 4;
  const int l16   = lane & 15;

  const bf16_t* W    = Wbt + (size_t)t * HSZ * HSZ;
  const float*  bias = (t == 0) ? bq : (t == 1) ? bk : bv;

  const int mw = m0 + wave * 32;                       // this wave's 32-row strip
  const int ak = lhalf * 8;                            // K sub-chunk per half
  const bf16_t* arow0 = Xb + (size_t)(mw + l16) * HSZ;
  const bf16_t* arow1 = Xb + (size_t)(mw + 16 + l16) * HSZ;

  v8f acc[2][4];
#pragma unroll
  for (int s = 0; s < 2; ++s)
#pragma unroll
    for (int i = 0; i < 4; ++i)
      acc[s][i] = (v8f){0.f, 0.f, 0.f, 0.f, 0.f, 0.f, 0.f, 0.f};

  for (int k0 = 0; k0 < HSZ; k0 += 32) {
    F16U a0, a1;
    a0.h[0] = *reinterpret_cast<const v8bf*>(arow0 + k0 + ak);
    a0.h[1] = *reinterpret_cast<const v8bf*>(arow0 + k0 + ak + 16);
    a1.h[0] = *reinterpret_cast<const v8bf*>(arow1 + k0 + ak);
    a1.h[1] = *reinterpret_cast<const v8bf*>(arow1 + k0 + ak + 16);
    const bf16_t* brow = W + (size_t)k0 * HSZ + n0 + l16;  // B-frag: lane = N col
    if (k0 + 32 < HSZ) {                     // WGP-scope prefetch of next K-step
      __builtin_prefetch(arow0 + k0 + 32, 0, 3);
      __builtin_prefetch(brow + 32 * HSZ, 0, 3);
    }
#pragma unroll
    for (int tt = 0; tt < 4; ++tt) {
      F16U b;
#pragma unroll
      for (int s = 0; s < 16; ++s)
        b.v[s] = brow[(size_t)(ak + (s & 7) + ((s >> 3) << 4)) * HSZ + tt * 16];
      acc[0][tt] = wmma_bf16(a0.v, b.v, acc[0][tt]);
      acc[1][tt] = wmma_bf16(a1.v, b.v, acc[1][tt]);
    }
  }

  // epilogue: bias add, bf16 convert, scatter per C-matrix layout
#pragma unroll
  for (int tt = 0; tt < 4; ++tt) {
    const int   n   = n0 + tt * 16 + l16;
    const float bv_ = bias[n];
    const int   hh  = n >> 6;
    const int   dd  = n & 63;
#pragma unroll
    for (int s = 0; s < 2; ++s) {
#pragma unroll
      for (int j = 0; j < 8; ++j) {
        const int r = mw + s * 16 + j + 8 * lhalf;    // global M row
        const int b = r >> 12;
        const int srow = r & 4095;
        const bf16_t o = (bf16_t)(acc[s][tt][j] + bv_);
        if (t == 1) {
          Kt[((size_t)(b * HH + hh) * DD + dd) * SS + srow] = o;
        } else {
          bf16_t* dst = (t == 0) ? Q : V;
          dst[((size_t)(b * HH + hh) * SS + srow) * DD + dd] = o;
        }
      }
    }
  }
}

// ---------------------------------------------------------------------------
// Kernel 4: BigBird attention, flash-style online softmax.
// grid = (64 qblocks, 16 heads, 4 batch), block = 128 (4 waves x 16 qrows).
// K/V tiles are DMA'd into LDS by the Tensor Data Mover (double-buffered,
// TENSORcnt-tracked), WMMA B-frags gather K/V from LDS per element.
// ---------------------------------------------------------------------------
__global__ __launch_bounds__(128) void bigbird_attn_kernel(
    const bf16_t* __restrict__ Q, const bf16_t* __restrict__ Kt,
    const bf16_t* __restrict__ V, float* __restrict__ Out) {
  __shared__ alignas(16) bf16_t kbuf[2][64 * 64];   // [d][key] tiles (8KB each)
  __shared__ alignas(16) bf16_t vbuf[2][64 * 64];   // [key][d] tiles (8KB each)
  __shared__ alignas(16) bf16_t lds_p[4][16 * 64];  // per-wave P tiles (8KB)

  const int l  = blockIdx.x;             // query block 0..63
  const int h  = blockIdx.y;
  const int b  = blockIdx.z;
  const int bh = b * HH + h;
  const int wave = threadIdx.x >> 5;
  const int lane = threadIdx.x & 31;
  const int lhalf = lane >> 4;
  const int l16   = lane & 15;
  const int ak    = lhalf * 8;

  const bf16_t* Qbh  = Q  + (size_t)bh * SS * DD;
  const bf16_t* Ktbh = Kt + (size_t)bh * DD * SS;
  const bf16_t* Vbh  = V  + (size_t)bh * SS * DD;

  const int q0 = l * 64 + wave * 16;

  // Q A-fragments, two K=32 chunks covering D=64
  const bf16_t* qrow = Qbh + (size_t)(q0 + l16) * DD;
  F16U qa[2];
#pragma unroll
  for (int c = 0; c < 2; ++c) {
    qa[c].h[0] = *reinterpret_cast<const v8bf*>(qrow + c * 32 + ak);
    qa[c].h[1] = *reinterpret_cast<const v8bf*>(qrow + c * 32 + ak + 16);
  }

  v8f oacc[4];
#pragma unroll
  for (int i = 0; i < 4; ++i) oacc[i] = (v8f){0.f, 0.f, 0.f, 0.f, 0.f, 0.f, 0.f, 0.f};
  float mrun[8], lrun[8];
#pragma unroll
  for (int j = 0; j < 8; ++j) { mrun[j] = -INFINITY; lrun[j] = 0.f; }

  const int nkb = (l == 0 || l == NB - 1) ? NB : ((l == 1 || l == NB - 2) ? 4 : 5);

#if HAVE_TDM
  // prologue: DMA block 0's K/V tiles into buffer 0
  if (wave == 0) {
    const int kb0 = kb_of(l, 0);
    tdm_load_2d_bf16((unsigned)(uintptr_t)&kbuf[0][0], Ktbh + kb0 * 64, 64, 64, SS);
    tdm_load_2d_bf16((unsigned)(uintptr_t)&vbuf[0][0], Vbh + (size_t)kb0 * 64 * DD, 64, 64, DD);
  }
#endif

  for (int i = 0; i < nkb; ++i) {
    const int buf = i & 1;

#if HAVE_TDM
    if (wave == 0) {
      if (i + 1 < nkb) {                       // DMA next tiles into other buffer
        const int kbn = kb_of(l, i + 1);
        tdm_load_2d_bf16((unsigned)(uintptr_t)&kbuf[buf ^ 1][0], Ktbh + kbn * 64, 64, 64, SS);
        tdm_load_2d_bf16((unsigned)(uintptr_t)&vbuf[buf ^ 1][0], Vbh + (size_t)kbn * 64 * DD,
                         64, 64, DD);
        __builtin_amdgcn_s_wait_tensorcnt(2); // this block's 2 DMAs done (in-order)
      } else {
        __builtin_amdgcn_s_wait_tensorcnt(0);
      }
    }
    __syncthreads();                           // publish LDS tiles to all waves
#else
    { // cooperative staging fallback (no TDM builtin on this toolchain)
      const int kb = kb_of(l, i);
      for (int idx = threadIdx.x; idx < (64 * 64) / 8; idx += 128) {
        const int row = (idx * 8) >> 6, col = (idx * 8) & 63;
        *reinterpret_cast<v8bf*>(&kbuf[buf][idx * 8]) =
            *reinterpret_cast<const v8bf*>(Ktbh + (size_t)row * SS + kb * 64 + col);
        *reinterpret_cast<v8bf*>(&vbuf[buf][idx * 8]) =
            *reinterpret_cast<const v8bf*>(Vbh + (size_t)kb * 64 * DD + idx * 8);
      }
      __syncthreads();
    }
#endif

    const bf16_t* ktile = &kbuf[buf][0];
    const bf16_t* vtile = &vbuf[buf][0];

    // ---- scores S = Q * K^T (16 x 64), f32 accumulate, B-frags from LDS ----
    v8f sacc[4];
#pragma unroll
    for (int tt = 0; tt < 4; ++tt) sacc[tt] = (v8f){0.f, 0.f, 0.f, 0.f, 0.f, 0.f, 0.f, 0.f};
#pragma unroll
    for (int c = 0; c < 2; ++c) {
      const bf16_t* krow = ktile + (size_t)(c * 32) * 64 + l16;   // B-frag: lane = N col (key)
#pragma unroll
      for (int tt = 0; tt < 4; ++tt) {
        F16U kf;
#pragma unroll
        for (int s = 0; s < 16; ++s)
          kf.v[s] = krow[(ak + (s & 7) + ((s >> 3) << 4)) * 64 + tt * 16];
        sacc[tt] = wmma_bf16(qa[c].v, kf.v, sacc[tt]);
      }
    }

    // ---- online softmax update, write P (bf16) to LDS ----
    float corr[8];
#pragma unroll
    for (int j = 0; j < 8; ++j) {
      float rm = -INFINITY;
#pragma unroll
      for (int tt = 0; tt < 4; ++tt) {
        float s = sacc[tt][j] * ATT_SCALE;
        sacc[tt][j] = s;
        rm = fmaxf(rm, s);
      }
      rm = rmax16(rm);
      const float mnew = fmaxf(mrun[j], rm);
      corr[j] = __expf(mrun[j] - mnew);
      float rsum = 0.f;
      const int prow_off = (j + 8 * lhalf) * 64 + l16;
#pragma unroll
      for (int tt = 0; tt < 4; ++tt) {
        float p = __expf(sacc[tt][j] - mnew);
        rsum += p;
        lds_p[wave][prow_off + tt * 16] = (bf16_t)p;
      }
      rsum = rsum16(rsum);
      lrun[j] = lrun[j] * corr[j] + rsum;
      mrun[j] = mnew;
    }

    // rescale running context by exp(m_old - m_new)
#pragma unroll
    for (int tt = 0; tt < 4; ++tt)
#pragma unroll
      for (int j = 0; j < 8; ++j) oacc[tt][j] *= corr[j];

    __builtin_amdgcn_wave_barrier();   // LDS P store -> A-frag load (same wave)

    // ---- re-layout P as A-fragments from LDS ----
    F16U pa[2];
    const bf16_t* prow = &lds_p[wave][l16 * 64];
#pragma unroll
    for (int c = 0; c < 2; ++c) {
      pa[c].h[0] = *reinterpret_cast<const v8bf*>(prow + c * 32 + ak);
      pa[c].h[1] = *reinterpret_cast<const v8bf*>(prow + c * 32 + ak + 16);
    }

    // ---- context += P * V, B-frags from LDS ----
#pragma unroll
    for (int c = 0; c < 2; ++c) {
      const bf16_t* vrow = vtile + (size_t)(c * 32) * 64 + l16;   // lane = N col (d)
#pragma unroll
      for (int tt = 0; tt < 4; ++tt) {
        F16U vf;
#pragma unroll
        for (int s = 0; s < 16; ++s)
          vf.v[s] = vrow[(ak + (s & 7) + ((s >> 3) << 4)) * 64 + tt * 16];
        oacc[tt] = wmma_bf16(pa[c].v, vf.v, oacc[tt]);
      }
    }

    __syncthreads();   // all waves done reading tiles before buffer reuse
  }

  // ---- finalize: divide by softmax denominator, store fp32 output ----
  float inv[8];
#pragma unroll
  for (int j = 0; j < 8; ++j) inv[j] = 1.0f / lrun[j];

#pragma unroll
  for (int tt = 0; tt < 4; ++tt) {
    const int d = tt * 16 + l16;
#pragma unroll
    for (int j = 0; j < 8; ++j) {
      const int s = q0 + j + 8 * lhalf;
      Out[((size_t)b * SS + s) * HSZ + h * 64 + d] = oacc[tt][j] * inv[j];
    }
  }
}

// ---------------------------------------------------------------------------
// Host launcher
// ---------------------------------------------------------------------------
extern "C" void kernel_launch(void* const* d_in, const int* in_sizes, int n_in,
                              void* d_out, int out_size, void* d_ws, size_t ws_size,
                              hipStream_t stream) {
  (void)in_sizes; (void)n_in; (void)out_size; (void)ws_size;
  const float* X  = (const float*)d_in[0];
  const float* Wq = (const float*)d_in[1];
  const float* bq = (const float*)d_in[2];
  const float* Wk = (const float*)d_in[3];
  const float* bk = (const float*)d_in[4];
  const float* Wv = (const float*)d_in[5];
  const float* bv = (const float*)d_in[6];
  float* Out = (float*)d_out;

  char* ws = (char*)d_ws;
  const size_t XB_BYTES  = (size_t)BB * SS * HSZ * 2;       // 32 MB
  const size_t WBT_BYTES = (size_t)3 * HSZ * HSZ * 2;       // 6 MB
  const size_t QKV_BYTES = (size_t)BB * HH * SS * DD * 2;   // 32 MB each
  bf16_t* Xb  = (bf16_t*)(ws);
  bf16_t* Wbt = (bf16_t*)(ws + XB_BYTES);
  bf16_t* Qb  = (bf16_t*)(ws + XB_BYTES + WBT_BYTES);
  bf16_t* Ktb = (bf16_t*)(ws + XB_BYTES + WBT_BYTES + QKV_BYTES);
  bf16_t* Vb  = (bf16_t*)(ws + XB_BYTES + WBT_BYTES + 2 * QKV_BYTES);

  // 1) X -> bf16
  {
    int n8 = (BB * SS * HSZ) / 8;                  // 2,097,152
    convert_x_kernel<<<(n8 + 255) / 256, 256, 0, stream>>>(X, Xb, n8);
  }
  // 2) W -> bf16, transposed to K-major
  {
    int n = 3 * HSZ * HSZ;                         // 3,145,728
    transpose_w_kernel<<<n / 256, 256, 0, stream>>>(Wq, Wk, Wv, Wbt);
  }
  // 3) fused QKV projection (M=16384, N=1024, K=1024), bf16 WMMA
  {
    dim3 grid((BB * SS) / 128, HSZ / 64, 3);       // (128,16,3)
    qkv_gemm_kernel<<<grid, 128, 0, stream>>>(Xb, Wbt, bq, bk, bv, Qb, Ktb, Vb);
  }
  // 4) block-sparse attention (TDM double-buffered K/V tiles)
  {
    dim3 grid(NB, HH, BB);                         // (64,16,4)
    bigbird_attn_kernel<<<grid, 128, 0, stream>>>(Qb, Ktb, Vb, Out);
  }
}
